// HSTRDA_86242943304140
// MI455X (gfx1250) — compile-verified, matrix operand order checked by reading, not run on hardware
//
#include <hip/hip_runtime.h>
#include <hip/hip_bf16.h>
#include <math.h>

typedef __attribute__((ext_vector_type(16))) __bf16 v16bf;
typedef __attribute__((ext_vector_type(8)))  float  v8f;

#define BNF   0.9999950000374997f   /* 1/sqrt(1+1e-5) */
#define BN2F  0.9999900000999990f   /* BNF^2           */
#define EPSF  1e-5f

__device__ __forceinline__ unsigned short f2bfu(float f){
  unsigned u = __builtin_bit_cast(unsigned, f);
  u += 0x7fffu + ((u >> 16) & 1u);          // round-to-nearest-even
  return (unsigned short)(u >> 16);
}
__device__ __forceinline__ __bf16 u2bf(unsigned short s){
  return __builtin_bit_cast(__bf16, s);
}
__device__ __forceinline__ float eluf(float x){ return x > 0.f ? x : (expf(x) - 1.f); }
__device__ __forceinline__ float geluf(float x){ return 0.5f * x * (1.f + erff(x * 0.70710678118654752f)); }

#define WMMA_BF16(a,b,c) __builtin_amdgcn_wmma_f32_16x16x32_bf16(false,(a),false,(b),(short)0,(c),false,false)

// A fragment (16x32 bf16, row-major in LDS, stride ld): lane = M, element e <-> K
// per ISA 7.12.2: laneLo: e<8 -> K=e, e>=8 -> K=e+8 ; laneHi adds +8.
__device__ __forceinline__ v16bf frag_rowmajor(const unsigned short* base, int ld, int kbase){
  int lane = threadIdx.x & 31;
  int m  = lane & 15;
  int ho = (lane >> 4) ? 8 : 0;
  v16bf v;
#pragma unroll
  for (int e = 0; e < 16; ++e){
    int k = kbase + (e < 8 ? e : e + 8) + ho;
    v[e] = u2bf(base[m * ld + k]);
  }
  return v;
}
// B fragment (32x16 bf16, B[k][n] stored row-major w[k*ld + n0+n]): lane = N, mirrored K map.
__device__ __forceinline__ v16bf frag_colB(const unsigned short* w, int ld, int kbase, int n0){
  int lane = threadIdx.x & 31;
  int n  = lane & 15;
  int ho = (lane >> 4) ? 8 : 0;
  v16bf v;
#pragma unroll
  for (int e = 0; e < 16; ++e){
    int k = kbase + (e < 8 ? e : e + 8) + ho;
    v[e] = u2bf(w[k * ld + n0 + n]);
  }
  return v;
}

/* ---------------- covariance: cov = Xc Xc^T / (T-1), 22x22 padded to 32x32,
   WMMA bf16 16x16x32, per-batch block, K streamed through LDS ------------- */
__global__ void cov_kernel(const float* __restrict__ x, float* __restrict__ cov){
  __shared__ float partial[22][8];
  __shared__ float meanL[22];
  __shared__ unsigned short xc[32 * 64];
  __shared__ float tbuf[8][256];
  int b = blockIdx.x, tid = threadIdx.x;
  const float* xb = x + (size_t)b * 22 * 16000;
  if (tid < 176){
    int ch = tid >> 3, p = tid & 7;
    float s = 0.f;
    for (int t = p; t < 16000; t += 8) s += xb[ch * 16000 + t];
    partial[ch][p] = s;
  }
  __syncthreads();
  if (tid < 22){
    float s = 0.f;
    for (int p = 0; p < 8; ++p) s += partial[tid][p];
    meanL[tid] = s * (1.f / 16000.f);
  }
  __syncthreads();
  int w = tid >> 5, lane = tid & 31;
  int tile = w & 3, r = tile >> 1, cc = tile & 1, kb = (w >> 2) * 32;
  v8f acc = {};
  for (int t0 = 0; t0 < 16000; t0 += 64){
    __syncthreads();
    for (int i = tid; i < 2048; i += 256){
      int row = i >> 6, t = i & 63;
      float v = (row < 22) ? (xb[row * 16000 + t0 + t] - meanL[row]) : 0.f;
      xc[i] = f2bfu(v);
    }
    if (t0 + 64 < 16000) __builtin_prefetch(xb + t0 + 64, 0, 1);
    __syncthreads();
    v16bf a  = frag_rowmajor(xc + r  * 16 * 64, 64, kb);
    v16bf bt = frag_rowmajor(xc + cc * 16 * 64, 64, kb);  // B[k][n] = Xc[c16+n][k]
    acc = WMMA_BF16(a, bt, acc);
  }
#pragma unroll
  for (int e = 0; e < 8; ++e) tbuf[w][e * 32 + lane] = acc[e];
  __syncthreads();
  if (w < 4){
    int ho = (lane >> 4) ? 8 : 0, n = lane & 15;
#pragma unroll
    for (int e = 0; e < 8; ++e){
      float v = (tbuf[w][e * 32 + lane] + tbuf[w + 4][e * 32 + lane]) * (1.f / 15999.f);
      int i = r * 16 + e + ho, j = cc * 16 + n;
      cov[(size_t)b * 1024 + i * 32 + j] = v;
    }
  }
}

__global__ void trace_kernel(const float* __restrict__ cov, float* __restrict__ trsum){
  __shared__ float tp[16];
  int tid = threadIdx.x;
  if (tid < 16){
    float s = 0.f;
    for (int i = 0; i < 22; ++i) s += cov[(size_t)tid * 1024 + i * 32 + i];
    tp[tid] = s + 22.f * 1e-5f;
  }
  __syncthreads();
  if (tid == 0){
    float s = 0.f;
    for (int i = 0; i < 16; ++i) s += tp[i];
    trsum[0] = s;
  }
}

/* -------- Cholesky 22x22 + log-diag + triu flatten (only diag nonzero) ---- */
__global__ void chol_kernel(const float* __restrict__ cov, const float* __restrict__ trsum,
                            float* __restrict__ flat){
  __shared__ float A[22][23];
  __shared__ float dlog[22];
  int b = blockIdx.x, tid = threadIdx.x;
  float dadd = 1e-5f + (trsum[0] * (1.f / 16.f)) * 0.001f;
  for (int i = tid; i < 484; i += 32){
    int r = i / 22, c = i % 22;
    float v = cov[(size_t)b * 1024 + r * 32 + c];
    if (r == c) v += dadd;
    A[r][c] = v;
  }
  __syncthreads();
  for (int k = 0; k < 22; ++k){
    if (tid == 0){
      float s = A[k][k];
      for (int p = 0; p < k; ++p) s -= A[k][p] * A[k][p];
      A[k][k] = sqrtf(fmaxf(s, 1e-30f));
    }
    __syncthreads();
    if (tid > k && tid < 22){
      float s = A[tid][k];
      for (int p = 0; p < k; ++p) s -= A[tid][p] * A[k][p];
      A[tid][k] = s / A[k][k];
    }
    __syncthreads();
  }
  if (tid < 22) dlog[tid] = logf(fmaxf(A[tid][tid], 1e-10f));
  __syncthreads();
  float* fb = flat + b * 253;
  for (int idx = tid; idx < 253; idx += 32) fb[idx] = 0.f;
  __syncthreads();
  if (tid < 22){
    int off = tid * 22 - (tid * (tid - 1)) / 2;   // triu row start, j==i
    fb[off] = A[tid][tid] * dlog[tid];
  }
}

/* ---------------- Riemannian MLP 253 -> 256 (elu) -> 48 ------------------- */
__global__ void riem_mlp(const float* __restrict__ flat, const float* __restrict__ w1,
                         const float* __restrict__ b1, const float* __restrict__ w2,
                         const float* __restrict__ b2, float* __restrict__ rfeat){
  __shared__ float fl[253];
  __shared__ float hid[256];
  int b = blockIdx.x, tid = threadIdx.x;
  if (tid < 253) fl[tid] = flat[b * 253 + tid];
  __syncthreads();
  float s = b1[tid];
  for (int i = 0; i < 253; ++i) s += fl[i] * w1[i * 256 + tid];
  hid[tid] = eluf(s);
  __syncthreads();
  if (tid < 48){
    float o = b2[tid];
    for (int j = 0; j < 256; ++j) o += hid[j] * w2[j * 48 + tid];
    rfeat[b * 48 + tid] = o;
  }
}

/* -------- reassociated conv front: electrode-mix (22->32) then per-channel
   64-tap conv + elu + avgpool8   (8x fewer FLOPs than conv1-then-dw1) ------ */
__global__ void convfront(const float* __restrict__ x, const float* __restrict__ c1w,
                          const float* __restrict__ d1w, float* __restrict__ out2){
  __shared__ float xload[22 * 256];  // 255 cols used, stride 256
  __shared__ float xm[32 * 256];
  __shared__ float dw[32 * 22];
  __shared__ float w1s[16 * 64];
  int b = blockIdx.x, ch = blockIdx.y, tid = threadIdx.x;
  int tp0 = ch * 24;
  const float* xb = x + (size_t)b * 22 * 16000;
  for (int i = tid; i < 704; i += 256)  dw[i]  = d1w[i];
  for (int i = tid; i < 1024; i += 256) w1s[i] = c1w[i];
  for (int i = tid; i < 22 * 255; i += 256){
    int r = i / 255, c = i % 255;
    int xi = 8 * tp0 - 32 + c;
    xload[r * 256 + c] = (xi >= 0 && xi < 16000) ? xb[r * 16000 + xi] : 0.f;
  }
  __builtin_prefetch(xb + 8 * tp0 + 224, 0, 1);
  __syncthreads();
  for (int i = tid; i < 32 * 255; i += 256){
    int c = i / 255, col = i % 255;
    float s = 0.f;
    for (int h = 0; h < 22; ++h) s += dw[c * 22 + h] * xload[h * 256 + col];
    xm[c * 256 + col] = s;
  }
  __syncthreads();
  for (int i = tid; i < 768; i += 256){
    int c = i / 24, ii = i % 24;
    int tp = tp0 + ii;
    if (tp >= 2000) continue;
    const float* wf  = w1s + (c >> 1) * 64;
    const float* xmc = xm + c * 256 + ii * 8;
    float acc = 0.f;
    for (int j = 0; j < 8; ++j){
      float s = 0.f;
      for (int k = 0; k < 64; ++k) s += wf[k] * xmc[j + k];
      acc += eluf(BN2F * s);
    }
    out2[((size_t)b * 32 + c) * 2000 + tp] = acc * 0.125f;
  }
}

/* -------- dw2 (16-tap, pad 8) + pointwise 32->48 + elu + avgpool8 --------- */
__global__ void dw2pw(const float* __restrict__ out2, const float* __restrict__ d2w,
                      const float* __restrict__ pww, float* __restrict__ feat){
  __shared__ float y3[32 * 400];
  __shared__ float pw[48 * 32];
  __shared__ float dwv[32 * 16];
  int b = blockIdx.x, ch = blockIdx.y, tid = threadIdx.x;
  int tp0 = ch * 50;
  for (int i = tid; i < 1536; i += 256) pw[i]  = pww[i];
  for (int i = tid; i < 512;  i += 256) dwv[i] = d2w[i];
  __syncthreads();
  const float* ob = out2 + (size_t)b * 32 * 2000;
  for (int i = tid; i < 12800; i += 256){
    int c = i / 400, tt = i % 400;
    int t = 8 * tp0 + tt;
    float s = 0.f;
    for (int k = 0; k < 16; ++k){
      int ti = t + k - 8;
      if (ti >= 0 && ti < 2000) s += dwv[c * 16 + k] * ob[c * 2000 + ti];
    }
    y3[c * 400 + tt] = BNF * s;
  }
  __syncthreads();
  for (int i = tid; i < 2400; i += 256){
    int o = i / 50, nn = i % 50;
    int n = tp0 + nn;
    float acc = 0.f;
    for (int j = 0; j < 8; ++j){
      int tt = nn * 8 + j;
      float s = 0.f;
      for (int c = 0; c < 32; ++c) s += pw[o * 32 + c] * y3[c * 400 + tt];
      acc += eluf(BNF * s);
    }
    feat[((size_t)b * 250 + n) * 48 + o] = acc * 0.125f;
  }
}

/* ---- temporal refinement: grouped conv3 + gelu + 1x1 conv, residual ------ */
__global__ void tc_kernel(float* __restrict__ feat, const float* __restrict__ tc1w,
                          const float* __restrict__ tc1b, const float* __restrict__ tc2w,
                          const float* __restrict__ tc2b){
  __shared__ float t1[24 * 250];
  int b = blockIdx.x, tid = threadIdx.x;
  const float* fb = feat + (size_t)b * 250 * 48;
  for (int i = tid; i < 6000; i += 256){
    int g = i / 250, n = i % 250;
    float s = tc1b[g];
    for (int ii = 0; ii < 2; ++ii){
      int chn = 2 * g + ii;
      for (int r = 0; r < 3; ++r){
        int nn = n + r - 1;
        if (nn >= 0 && nn < 250) s += tc1w[(g * 2 + ii) * 3 + r] * fb[nn * 48 + chn];
      }
    }
    t1[g * 250 + n] = geluf(s * BNF);
  }
  __syncthreads();
  for (int i = tid; i < 12000; i += 256){
    int n = i / 48, o = i % 48;
    float s = tc2b[o];
    for (int g = 0; g < 24; ++g) s += tc2w[o * 24 + g] * t1[g * 250 + n];
    feat[(size_t)b * 250 * 48 + n * 48 + o] = fb[n * 48 + o] + s;
  }
}

/* -------- concat [r_feat; feat] @ fusion_proj + pos_enc ------------------- */
__global__ void fuse_kernel(const float* __restrict__ rfeat, const float* __restrict__ feat,
                            const float* __restrict__ fp, const float* __restrict__ pos,
                            float* __restrict__ fused){
  int gid = blockIdx.x * 256 + threadIdx.x;
  if (gid >= 16 * 251 * 48) return;
  int b = gid / (251 * 48), rem = gid % (251 * 48), n = rem / 48, o = rem % 48;
  const float* row = (n == 0) ? (rfeat + b * 48) : (feat + ((size_t)b * 250 + n - 1) * 48);
  float s = 0.f;
  for (int i = 0; i < 48; ++i) s += row[i] * fp[i * 48 + o];
  fused[gid] = s + pos[n * 48 + o];
}

/* -------- transformer: LN1 + QKV GEMM (48->144) via WMMA bf16 ------------- */
__global__ void ln_qkv(const float* __restrict__ fused, const float* __restrict__ g,
                       const float* __restrict__ be, const float* __restrict__ qw,
                       const float* __restrict__ qb, float* __restrict__ qkv){
  __shared__ unsigned short xl[16 * 64];
  __shared__ unsigned short wl[64 * 144];
  int b = blockIdx.x, rt = blockIdx.y, tid = threadIdx.x;
  if (tid < 16){
    int row = rt * 16 + tid;
    if (row < 251){
      const float* fr = fused + ((size_t)b * 251 + row) * 48;
      float m = 0.f; for (int k = 0; k < 48; ++k) m += fr[k]; m *= (1.f / 48.f);
      float v = 0.f; for (int k = 0; k < 48; ++k){ float d = fr[k] - m; v += d * d; } v *= (1.f / 48.f);
      float inv = rsqrtf(v + EPSF);
      for (int k = 0; k < 48; ++k) xl[tid * 64 + k] = f2bfu((fr[k] - m) * inv * g[k] + be[k]);
      for (int k = 48; k < 64; ++k) xl[tid * 64 + k] = 0;
    } else {
      for (int k = 0; k < 64; ++k) xl[tid * 64 + k] = 0;
    }
  }
  for (int i = tid; i < 64 * 144; i += 256){
    int k = i / 144, n = i % 144;
    wl[i] = (k < 48) ? f2bfu(qw[k * 144 + n]) : (unsigned short)0;
  }
  __syncthreads();
  int w = tid >> 5, lane = tid & 31;
  v16bf a0 = frag_rowmajor(xl, 64, 0);
  v16bf a1 = frag_rowmajor(xl, 64, 32);
  for (int nt = w; nt < 9; nt += 8){
    v16bf b0 = frag_colB(wl, 144, 0,  nt * 16);
    v16bf b1 = frag_colB(wl, 144, 32, nt * 16);
    v8f acc = {};
    acc = WMMA_BF16(a0, b0, acc);
    acc = WMMA_BF16(a1, b1, acc);
    int ho = (lane >> 4) ? 8 : 0, n = lane & 15, col = nt * 16 + n;
    float bias = qb[col];
#pragma unroll
    for (int e = 0; e < 8; ++e){
      int row = rt * 16 + e + ho;
      if (row < 251) qkv[((size_t)b * 251 + row) * 144 + col] = acc[e] + bias;
    }
  }
}

/* -------- banded attention (win=5, HD=6): tiny K, VALU is optimal --------- */
__global__ void attn_kernel(const float* __restrict__ qkv, float* __restrict__ ob){
  int gid = blockIdx.x * 256 + threadIdx.x;
  if (gid >= 16 * 8 * 251) return;
  int b = gid / (8 * 251), r = gid % (8 * 251), h = r / 251, q = r % 251;
  const float* base = qkv + (size_t)b * 251 * 144;
  float qv[6];
  for (int d = 0; d < 6; ++d) qv[d] = base[q * 144 + h * 6 + d];
  int k0 = q - 5 < 0 ? 0 : q - 5, k1 = q + 5 > 250 ? 250 : q + 5;
  float sc[11]; float mx = -1e30f;
  for (int kk = k0; kk <= k1; ++kk){
    const float* kp = base + kk * 144 + 48 + h * 6;
    float s = 0.f;
    for (int d = 0; d < 6; ++d) s += qv[d] * kp[d];
    s *= 0.40824829046386302f;  // 6^-0.5
    sc[kk - k0] = s; mx = fmaxf(mx, s);
  }
  float sum = 0.f, o[6] = {0, 0, 0, 0, 0, 0};
  for (int kk = k0; kk <= k1; ++kk){
    float p = expf(sc[kk - k0] - mx);
    sum += p;
    const float* vp = base + kk * 144 + 96 + h * 6;
    for (int d = 0; d < 6; ++d) o[d] += p * vp[d];
  }
  float inv = 1.f / sum;
  float* op = ob + ((size_t)b * 251 + q) * 48 + h * 6;
  for (int d = 0; d < 6; ++d) op[d] = o[d] * inv;
}

/* -------- attention projection 48->48 (WMMA) + residual ------------------- */
__global__ void proj_res(const float* __restrict__ obuf, const float* __restrict__ pw,
                         const float* __restrict__ pb, float* __restrict__ fused){
  __shared__ unsigned short xl[16 * 64];
  __shared__ unsigned short wl[64 * 48];
  int b = blockIdx.x, rt = blockIdx.y, tid = threadIdx.x;
  for (int i = tid; i < 16 * 64; i += 256){
    int m = i >> 6, k = i & 63;
    int row = rt * 16 + m;
    xl[i] = (row < 251 && k < 48) ? f2bfu(obuf[((size_t)b * 251 + row) * 48 + k]) : (unsigned short)0;
  }
  for (int i = tid; i < 64 * 48; i += 256){
    int k = i / 48, n = i % 48;
    wl[i] = (k < 48) ? f2bfu(pw[k * 48 + n]) : (unsigned short)0;
  }
  __syncthreads();
  int w = tid >> 5, lane = tid & 31;
  if (w < 3){
    v16bf a0 = frag_rowmajor(xl, 64, 0);
    v16bf a1 = frag_rowmajor(xl, 64, 32);
    v16bf b0 = frag_colB(wl, 48, 0,  w * 16);
    v16bf b1 = frag_colB(wl, 48, 32, w * 16);
    v8f acc = {};
    acc = WMMA_BF16(a0, b0, acc);
    acc = WMMA_BF16(a1, b1, acc);
    int ho = (lane >> 4) ? 8 : 0, n = lane & 15, col = w * 16 + n;
    float bias = pb[col];
#pragma unroll
    for (int e = 0; e < 8; ++e){
      int row = rt * 16 + e + ho;
      if (row < 251){
        size_t idx = ((size_t)b * 251 + row) * 48 + col;
        fused[idx] = fused[idx] + acc[e] + bias;
      }
    }
  }
}

/* -------- LN2 + FFN1 48->192 (WMMA) + exact gelu -------------------------- */
__global__ void ln2_ffn1(const float* __restrict__ fused, const float* __restrict__ g,
                         const float* __restrict__ be, const float* __restrict__ fw,
                         const float* __restrict__ fbias, float* __restrict__ hbuf){
  __shared__ unsigned short xl[16 * 64];
  __shared__ unsigned short wl[64 * 192];
  int b = blockIdx.x, rt = blockIdx.y, tid = threadIdx.x;
  if (tid < 16){
    int row = rt * 16 + tid;
    if (row < 251){
      const float* fr = fused + ((size_t)b * 251 + row) * 48;
      float m = 0.f; for (int k = 0; k < 48; ++k) m += fr[k]; m *= (1.f / 48.f);
      float v = 0.f; for (int k = 0; k < 48; ++k){ float d = fr[k] - m; v += d * d; } v *= (1.f / 48.f);
      float inv = rsqrtf(v + EPSF);
      for (int k = 0; k < 48; ++k) xl[tid * 64 + k] = f2bfu((fr[k] - m) * inv * g[k] + be[k]);
      for (int k = 48; k < 64; ++k) xl[tid * 64 + k] = 0;
    } else {
      for (int k = 0; k < 64; ++k) xl[tid * 64 + k] = 0;
    }
  }
  for (int i = tid; i < 64 * 192; i += 256){
    int k = i / 192, n = i % 192;
    wl[i] = (k < 48) ? f2bfu(fw[k * 192 + n]) : (unsigned short)0;
  }
  __syncthreads();
  int w = tid >> 5, lane = tid & 31;
  v16bf a0 = frag_rowmajor(xl, 64, 0);
  v16bf a1 = frag_rowmajor(xl, 64, 32);
  for (int nt = w; nt < 12; nt += 8){
    v16bf b0 = frag_colB(wl, 192, 0,  nt * 16);
    v16bf b1 = frag_colB(wl, 192, 32, nt * 16);
    v8f acc = {};
    acc = WMMA_BF16(a0, b0, acc);
    acc = WMMA_BF16(a1, b1, acc);
    int ho = (lane >> 4) ? 8 : 0, n = lane & 15, col = nt * 16 + n;
    float bias = fbias[col];
#pragma unroll
    for (int e = 0; e < 8; ++e){
      int row = rt * 16 + e + ho;
      if (row < 251){
        float v = acc[e] + bias;
        hbuf[((size_t)b * 251 + row) * 192 + col] = geluf(v);
      }
    }
  }
}

/* -------- FFN2 192->48 (WMMA, K=6x32) + residual -------------------------- */
__global__ void ffn2_res(const float* __restrict__ hbuf, const float* __restrict__ fw,
                         const float* __restrict__ fbias, float* __restrict__ fused){
  __shared__ unsigned short xl[16 * 192];
  __shared__ unsigned short wl[192 * 48];
  int b = blockIdx.x, rt = blockIdx.y, tid = threadIdx.x;
  for (int i = tid; i < 16 * 192; i += 256){
    int m = i / 192, k = i % 192;
    int row = rt * 16 + m;
    xl[i] = (row < 251) ? f2bfu(hbuf[((size_t)b * 251 + row) * 192 + k]) : (unsigned short)0;
  }
  for (int i = tid; i < 192 * 48; i += 256) wl[i] = f2bfu(fw[i]);
  __syncthreads();
  int w = tid >> 5, lane = tid & 31;
  if (w < 3){
    v8f acc = {};
#pragma unroll
    for (int kb = 0; kb < 6; ++kb){
      v16bf a  = frag_rowmajor(xl, 192, kb * 32);
      v16bf bb = frag_colB(wl, 48, kb * 32, w * 16);
      acc = WMMA_BF16(a, bb, acc);
    }
    int ho = (lane >> 4) ? 8 : 0, n = lane & 15, col = w * 16 + n;
    float bias = fbias[col];
#pragma unroll
    for (int e = 0; e < 8; ++e){
      int row = rt * 16 + e + ho;
      if (row < 251){
        size_t idx = ((size_t)b * 251 + row) * 48 + col;
        fused[idx] = fused[idx] + acc[e] + bias;
      }
    }
  }
}

/* -------- mean-pool over N, LN, classifier 48->4 -------------------------- */
__global__ void final_kernel(const float* __restrict__ fused, const float* __restrict__ cg,
                             const float* __restrict__ cb, const float* __restrict__ cw,
                             const float* __restrict__ cbias, float* __restrict__ out){
  __shared__ float gm[48];
  __shared__ float gn[48];
  __shared__ float stats[2];
  int b = blockIdx.x, tid = threadIdx.x;
  if (tid < 48){
    float s = 0.f;
    for (int n = 0; n < 251; ++n) s += fused[((size_t)b * 251 + n) * 48 + tid];
    gm[tid] = s * (1.f / 251.f);
  }
  __syncthreads();
  if (tid == 0){
    float m = 0.f; for (int d = 0; d < 48; ++d) m += gm[d]; m *= (1.f / 48.f);
    float v = 0.f; for (int d = 0; d < 48; ++d){ float x = gm[d] - m; v += x * x; } v *= (1.f / 48.f);
    stats[0] = m; stats[1] = rsqrtf(v + EPSF);
  }
  __syncthreads();
  if (tid < 48) gn[tid] = (gm[tid] - stats[0]) * stats[1] * cg[tid] + cb[tid];
  __syncthreads();
  if (tid < 4){
    float s = cbias[tid];
    for (int d = 0; d < 48; ++d) s += gn[d] * cw[d * 4 + tid];
    out[b * 4 + tid] = s;
  }
}

extern "C" void kernel_launch(void* const* d_in, const int* in_sizes, int n_in,
                              void* d_out, int out_size, void* d_ws, size_t ws_size,
                              hipStream_t stream){
  const float* x        = (const float*)d_in[0];
  const float* riem_w1  = (const float*)d_in[1];
  const float* riem_b1  = (const float*)d_in[2];
  const float* riem_w2  = (const float*)d_in[3];
  const float* riem_b2  = (const float*)d_in[4];
  const float* conv1_w  = (const float*)d_in[5];
  const float* dw1_w    = (const float*)d_in[6];
  const float* dw2_w    = (const float*)d_in[7];
  const float* pw_w     = (const float*)d_in[8];
  const float* tc1_w    = (const float*)d_in[9];
  const float* tc1_b    = (const float*)d_in[10];
  const float* tc2_w    = (const float*)d_in[11];
  const float* tc2_b    = (const float*)d_in[12];
  const float* fusion_p = (const float*)d_in[13];
  const float* pos_enc  = (const float*)d_in[14];
  const float* ln1_g    = (const float*)d_in[15];
  const float* ln1_b    = (const float*)d_in[16];
  const float* qkv_w    = (const float*)d_in[17];
  const float* qkv_b    = (const float*)d_in[18];
  const float* apw      = (const float*)d_in[19];
  const float* apb      = (const float*)d_in[20];
  const float* ln2_g    = (const float*)d_in[21];
  const float* ln2_b    = (const float*)d_in[22];
  const float* fw1      = (const float*)d_in[23];
  const float* fb1      = (const float*)d_in[24];
  const float* fw2      = (const float*)d_in[25];
  const float* fb2      = (const float*)d_in[26];
  const float* clg      = (const float*)d_in[27];
  const float* clb      = (const float*)d_in[28];
  const float* clw      = (const float*)d_in[29];
  const float* clbias   = (const float*)d_in[30];
  (void)in_sizes; (void)n_in; (void)out_size; (void)ws_size;

  float* ws    = (float*)d_ws;
  float* TR    = ws;                       // 1
  float* COV   = ws + 64;                  // 16 * 32*32
  float* FLAT  = COV  + 16 * 1024;         // 16 * 253
  float* RFEAT = FLAT + 16 * 253;          // 16 * 48
  float* FEAT  = RFEAT + 16 * 48;          // 16 * 250 * 48
  float* FUSED = FEAT  + 16 * 250 * 48;    // 16 * 251 * 48
  float* QKV   = FUSED + 16 * 251 * 48;    // 16 * 251 * 144
  float* OBUF  = QKV   + 16 * 251 * 144;   // 16 * 251 * 48
  float* HBUF  = OBUF  + 16 * 251 * 48;    // 16 * 251 * 192
  float* OUT2  = HBUF  + 16 * 251 * 192;   // 16 * 32 * 2000   (~11.9 MB total)

  cov_kernel  <<<16, 256, 0, stream>>>(x, COV);
  trace_kernel<<<1, 32, 0, stream>>>(COV, TR);
  chol_kernel <<<16, 32, 0, stream>>>(COV, TR, FLAT);
  riem_mlp    <<<16, 256, 0, stream>>>(FLAT, riem_w1, riem_b1, riem_w2, riem_b2, RFEAT);
  convfront   <<<dim3(16, 84), 256, 0, stream>>>(x, conv1_w, dw1_w, OUT2);
  dw2pw       <<<dim3(16, 5), 256, 0, stream>>>(OUT2, dw2_w, pw_w, FEAT);
  tc_kernel   <<<16, 256, 0, stream>>>(FEAT, tc1_w, tc1_b, tc2_w, tc2_b);
  int tot = 16 * 251 * 48;
  fuse_kernel <<<(tot + 255) / 256, 256, 0, stream>>>(RFEAT, FEAT, fusion_p, pos_enc, FUSED);
  for (int l = 0; l < 4; ++l){
    ln_qkv    <<<dim3(16, 16), 256, 0, stream>>>(FUSED, ln1_g + 48 * l, ln1_b + 48 * l,
                                                 qkv_w + l * 48 * 144, qkv_b + l * 144, QKV);
    int na = 16 * 8 * 251;
    attn_kernel<<<(na + 255) / 256, 256, 0, stream>>>(QKV, OBUF);
    proj_res  <<<dim3(16, 16), 256, 0, stream>>>(OBUF, apw + l * 48 * 48, apb + 48 * l, FUSED);
    ln2_ffn1  <<<dim3(16, 16), 256, 0, stream>>>(FUSED, ln2_g + 48 * l, ln2_b + 48 * l,
                                                 fw1 + l * 48 * 192, fb1 + 192 * l, HBUF);
    ffn2_res  <<<dim3(16, 16), 256, 0, stream>>>(HBUF, fw2 + l * 192 * 48, fb2 + 48 * l, FUSED);
  }
  final_kernel<<<16, 64, 0, stream>>>(FUSED, clg, clb, clw, clbias, (float*)d_out);
}
